// HGRNBitAttention_52733608460967
// MI455X (gfx1250) — 8-bit matrix operand order repaired and checked by reading, compiled, not run on hardware
//
#include <hip/hip_runtime.h>
#include <cstdint>
#include <cstddef>


typedef int v8i __attribute__((ext_vector_type(8)));
typedef int gvec4 __attribute__((vector_size(16)));

#if defined(__AMDGCN__) && __has_builtin(__builtin_amdgcn_global_load_async_to_lds_b128)
#define HGRN_ASYNC 1
#else
#define HGRN_ASYNC 0
#endif

namespace hgrn {

constexpr int BATCH = 4;
constexpr int Lseq  = 2048;
constexpr int D     = 2048;
constexpr int M     = BATCH * Lseq;
constexpr int Kd    = D;
constexpr int Nd    = D;
constexpr size_t DD = (size_t)D * D;
constexpr size_t MD = (size_t)M * D;

constexpr int CH = 16;
constexpr int CL = Lseq / CH;
constexpr int BD = BATCH * D;

constexpr int PITCH = 80;
constexpr int NK    = Kd / 64;

constexpr size_t OFF_WSUM  = 0;
constexpr size_t OFF_WPART = 1024;
constexpr size_t OFF_XSCL  = 8192;
constexpr size_t OFF_OSCL  = OFF_XSCL + (size_t)M * 4;
constexpr size_t OFF_WQ    = OFF_OSCL + (size_t)M * 4;
constexpr size_t OFF_XQ    = OFF_WQ + 4 * DD;
constexpr size_t OFF_OQ    = OFF_XQ + MD;
constexpr size_t OFF_IRAW  = OFF_OQ + MD;
constexpr size_t OFF_FRAW  = OFF_IRAW + MD * 4;
constexpr size_t OFF_GRAW  = OFF_FRAW + MD * 4;
constexpr size_t OFF_CSP   = OFF_GRAW + MD * 4;
constexpr size_t OFF_CSH   = OFF_CSP + (size_t)CH * BD * 4;
constexpr size_t OFF_CARRY = OFF_CSH + (size_t)CH * BD * 4;

union V8U {
  v8i v;
  long long l[4];
  int4 q[2];
};

__device__ __forceinline__ float sigmoidf_dev(float x) {
  return 1.0f / (1.0f + __expf(-x));
}

__device__ __forceinline__ float block_reduce_sum(float v, float* red) {
  const int tid = threadIdx.x;
  red[tid] = v;
  __syncthreads();
  for (int s = 128; s >= 1; s >>= 1) {
    if (tid < s) red[tid] += red[tid + s];
    __syncthreads();
  }
  float r = red[0];
  __syncthreads();
  return r;
}

__device__ __forceinline__ float block_reduce_max(float v, float* red) {
  const int tid = threadIdx.x;
  red[tid] = v;
  __syncthreads();
  for (int s = 128; s >= 1; s >>= 1) {
    if (tid < s) red[tid] = fmaxf(red[tid], red[tid + s]);
    __syncthreads();
  }
  float r = red[0];
  __syncthreads();
  return r;
}

__device__ __forceinline__ void wait_async0() {
#if HGRN_ASYNC
#if __has_builtin(__builtin_amdgcn_s_wait_asynccnt)
  __builtin_amdgcn_s_wait_asynccnt(0);
#else
  asm volatile("s_wait_asynccnt 0x0" ::: "memory");
#endif
#endif
}

__device__ __forceinline__ void stage_tile(const int8_t* __restrict__ gsrc,
                                           int8_t* sdst, int tid) {
  const int r0 = tid >> 2;
  const int c  = (tid & 3) * 16;
#pragma unroll
  for (int p = 0; p < 2; ++p) {
    const int row = r0 + p * 64;
    const int8_t* src = gsrc + (size_t)row * Kd + c;
    int8_t* dst = sdst + row * PITCH + c;
#if HGRN_ASYNC
    __builtin_amdgcn_global_load_async_to_lds_b128(
        (__attribute__((address_space(1))) gvec4*)(gvec4*)(void*)const_cast<int8_t*>(src),
        (__attribute__((address_space(3))) gvec4*)(gvec4*)(void*)dst, 0, 0);
#else
    *(int4*)dst = *(const int4*)src;
#endif
  }
}

__global__ __launch_bounds__(256) void wabs_partial_kernel(
    const float* __restrict__ w, float* __restrict__ partial) {
  __shared__ float red[256];
  const int tid = threadIdx.x;
  size_t idx = (size_t)blockIdx.x * 256 + tid;
  float s = 0.0f;
  for (; idx < DD; idx += 256 * 256) s += fabsf(w[idx]);
  s = block_reduce_sum(s, red);
  if (tid == 0) partial[blockIdx.x] = s;
}

__global__ __launch_bounds__(256) void wsum_final_kernel(
    const float* __restrict__ partial, float* __restrict__ wsum) {
  __shared__ float red[256];
  const int tid = threadIdx.x;
  float s = partial[(size_t)blockIdx.x * 256 + tid];
  s = block_reduce_sum(s, red);
  if (tid == 0) wsum[blockIdx.x] = s;
}

__global__ __launch_bounds__(256) void wquant_kernel(
    const float* __restrict__ w, const float* __restrict__ wsumPtr,
    int8_t* __restrict__ wq) {
  const float wmean = wsumPtr[0] * (1.0f / (float)DD);
  const float ws = 1.0f / fmaxf(wmean, 1e-5f);
  size_t i = (size_t)blockIdx.x * blockDim.x + threadIdx.x;
  const size_t stride = (size_t)gridDim.x * blockDim.x;
  for (; i < DD; i += stride) {
    float q = fminf(fmaxf(rintf(w[i] * ws), -1.0f), 1.0f);
    wq[i] = (int8_t)(int)q;
  }
}

__global__ __launch_bounds__(256) void act_quant_kernel(
    const float* __restrict__ x, int8_t* __restrict__ xq,
    float* __restrict__ xscale) {
  __shared__ float red[256];
  const int m = blockIdx.x;
  const int tid = threadIdx.x;
  const float* xr = x + (size_t)m * D;

  float ss = 0.0f;
  for (int d = tid; d < D; d += 256) { float v = xr[d]; ss += v * v; }
  ss = block_reduce_sum(ss, red);
  const float rn = rsqrtf(ss * (1.0f / (float)D) + 1e-5f);

  float am = 0.0f;
  for (int d = tid; d < D; d += 256) am = fmaxf(am, fabsf(xr[d] * rn));
  am = block_reduce_max(am, red);
  const float amc = fmaxf(am, 1e-5f);
  const float s = 127.0f / amc;

  for (int d = tid; d < D; d += 256) {
    float q = fminf(fmaxf(rintf(xr[d] * rn * s), -128.0f), 127.0f);
    xq[(size_t)m * D + d] = (int8_t)(int)q;
  }
  if (tid == 0) xscale[m] = amc * (1.0f / 127.0f);
}

__global__ __launch_bounds__(256, 2) void gemm_iu8_kernel(
    const int8_t* __restrict__ A, const int8_t* __restrict__ W,
    const float* __restrict__ ascale, const float* __restrict__ wsumPtr,
    float* __restrict__ out) {
  __shared__ int8_t sA[2][128 * PITCH];
  __shared__ int8_t sB[2][128 * PITCH];

  const int tid  = threadIdx.x;
  const int wv   = tid >> 5;
  const int lane = tid & 31;
  const int half = lane >> 4;
  const int l16  = lane & 15;
  const int waveMl = (wv >> 1) * 32;
  const int waveNl = (wv & 1) * 64;
  const int waveM  = blockIdx.y * 128 + waveMl;
  const int waveN  = blockIdx.x * 128 + waveNl;

  const int8_t* Ablk = A + (size_t)(blockIdx.y * 128) * Kd;
  const int8_t* Wblk = W + (size_t)(blockIdx.x * 128) * Kd;

  const float wdeq = fmaxf(wsumPtr[0] * (1.0f / (float)DD), 1e-5f);

  v8i acc[2][4];
#pragma unroll
  for (int mi = 0; mi < 2; ++mi)
#pragma unroll
    for (int ni = 0; ni < 4; ++ni) { v8i z = {}; acc[mi][ni] = z; }

#if HGRN_ASYNC
  stage_tile(Ablk, sA[0], tid);
  stage_tile(Wblk, sB[0], tid);
#endif

  for (int kt = 0; kt < NK; ++kt) {
    const int cur = kt & 1;
#if HGRN_ASYNC
    wait_async0();
    __syncthreads();
    if (kt + 1 < NK) {
      stage_tile(Ablk + (kt + 1) * 64, sA[(kt + 1) & 1], tid);
      stage_tile(Wblk + (kt + 1) * 64, sB[(kt + 1) & 1], tid);
    }
#else
    __syncthreads();
    stage_tile(Ablk + kt * 64, sA[cur], tid);
    stage_tile(Wblk + kt * 64, sB[cur], tid);
    __syncthreads();
#endif
    if (kt + 2 < NK) {
      __builtin_prefetch(Ablk + (size_t)(tid >> 2) * Kd + (kt + 2) * 64, 0, 1);
      __builtin_prefetch(Wblk + (size_t)(tid >> 2) * Kd + (kt + 2) * 64, 0, 1);
    }

    v8i a[2], b[4];
#pragma unroll
    for (int mi = 0; mi < 2; ++mi) {
      const int8_t* ap = &sA[cur][(waveMl + mi * 16 + l16) * PITCH + half * 16];
      V8U u;
      u.l[0] = *(const long long*)(ap + 0);
      u.l[1] = *(const long long*)(ap + 8);
      u.l[2] = *(const long long*)(ap + 32);
      u.l[3] = *(const long long*)(ap + 40);
      a[mi] = u.v;
    }
#pragma unroll
    for (int ni = 0; ni < 4; ++ni) {
      const int8_t* bp = &sB[cur][(waveNl + ni * 16 + l16) * PITCH + half * 16];
      V8U u;
      u.q[0] = *(const int4*)(bp + 0);
      u.q[1] = *(const int4*)(bp + 32);
      b[ni] = u.v;
    }
#pragma unroll
    for (int mi = 0; mi < 2; ++mi)
#pragma unroll
      for (int ni = 0; ni < 4; ++ni)
        acc[mi][ni] = __builtin_amdgcn_wmma_i32_16x16x64_iu8(
            true, a[mi], true, b[ni], acc[mi][ni], false, false);
  }

#pragma unroll
  for (int mi = 0; mi < 2; ++mi) {
#pragma unroll
    for (int r = 0; r < 8; ++r) {
      const int m = waveM + mi * 16 + half * 8 + r;
      const float sc = ascale[m] * wdeq;
#pragma unroll
      for (int ni = 0; ni < 4; ++ni) {
        const int n = waveN + ni * 16 + l16;
        out[(size_t)m * Nd + n] = (float)acc[mi][ni][r] * sc;
      }
    }
  }
}

__global__ __launch_bounds__(256) void swiglu_kernel(
    float* __restrict__ ib, float* __restrict__ fb) {
  size_t i = (size_t)blockIdx.x * blockDim.x + threadIdx.x;
  const size_t stride = (size_t)gridDim.x * blockDim.x;
  for (; i < MD; i += stride) {
    const float fr = fb[i];
    const float ir = ib[i];
    const float fs = sigmoidf_dev(fr);
    const float si = ir * sigmoidf_dev(ir);
    ib[i] = si * (1.0f - fs);
    fb[i] = fs;
  }
}

__global__ __launch_bounds__(256) void scan_pass1_kernel(
    float* __restrict__ hb, float* __restrict__ fb,
    float* __restrict__ csP, float* __restrict__ csH) {
  const int u = blockIdx.x * 256 + threadIdx.x;
  const int c = u >> 13;
  const int t = u & (BD - 1);
  const int b = t >> 11;
  const int d = t & (D - 1);
  const size_t base = ((size_t)b * Lseq + (size_t)c * CL) * D + d;
  float h = 0.0f, F = 1.0f;
  for (int l = 0; l < CL; ++l) {
    const size_t idx = base + (size_t)l * D;
    const float f = fb[idx];
    h = fmaf(f, h, hb[idx]);
    F *= f;
    hb[idx] = h;
    fb[idx] = F;
  }
  csP[(size_t)c * BD + t] = F;
  csH[(size_t)c * BD + t] = h;
}

__global__ __launch_bounds__(256) void scan_pass2_kernel(
    const float* __restrict__ csP, const float* __restrict__ csH,
    float* __restrict__ carry) {
  const int t = blockIdx.x * 256 + threadIdx.x;
  float c = 0.0f;
#pragma unroll
  for (int ck = 0; ck < CH; ++ck) {
    carry[(size_t)ck * BD + t] = c;
    c = fmaf(csP[(size_t)ck * BD + t], c, csH[(size_t)ck * BD + t]);
  }
}

__global__ __launch_bounds__(256) void scan_pass3_kernel(
    float* __restrict__ hb, const float* __restrict__ fb,
    const float* __restrict__ carry) {
  size_t i = (size_t)blockIdx.x * blockDim.x + threadIdx.x;
  const size_t stride = (size_t)gridDim.x * blockDim.x;
  for (; i < MD; i += stride) {
    const int b = (int)(i >> 22);
    const size_t rem = i & ((size_t)Lseq * D - 1);
    const int l = (int)(rem >> 11);
    const int d = (int)(rem & (D - 1));
    const int c = l >> 7;
    const int t = (b << 11) + d;
    hb[i] = fmaf(carry[(size_t)c * BD + t], fb[i], hb[i]);
  }
}

__global__ __launch_bounds__(256) void gate_quant_kernel(
    const float* __restrict__ g, const float* __restrict__ h,
    const float* __restrict__ gw, int8_t* __restrict__ oq,
    float* __restrict__ oscale) {
  __shared__ float osm[D];
  __shared__ float red[256];
  const int m = blockIdx.x;
  const int tid = threadIdx.x;
  const float* gr = g + (size_t)m * D;
  const float* hr = h + (size_t)m * D;

  float ssg = 0.0f;
  for (int d = tid; d < D; d += 256) { float v = gr[d]; ssg += v * v; }
  ssg = block_reduce_sum(ssg, red);
  const float rng = rsqrtf(ssg * (1.0f / (float)D) + 1e-5f);

  float sso = 0.0f;
  for (int d = tid; d < D; d += 256) {
    const float hv = hr[d];
    const float o = gr[d] * rng * gw[d] * (hv * sigmoidf_dev(hv));
    osm[d] = o;
    sso += o * o;
  }
  sso = block_reduce_sum(sso, red);
  const float rno = rsqrtf(sso * (1.0f / (float)D) + 1e-5f);

  float am = 0.0f;
  for (int d = tid; d < D; d += 256) am = fmaxf(am, fabsf(osm[d] * rno));
  am = block_reduce_max(am, red);
  const float amc = fmaxf(am, 1e-5f);
  const float s = 127.0f / amc;

  for (int d = tid; d < D; d += 256) {
    float q = fminf(fmaxf(rintf(osm[d] * rno * s), -128.0f), 127.0f);
    oq[(size_t)m * D + d] = (int8_t)(int)q;
  }
  if (tid == 0) oscale[m] = amc * (1.0f / 127.0f);
}

}

extern "C" void kernel_launch(void* const* d_in, const int* in_sizes, int n_in,
                              void* d_out, int out_size, void* d_ws, size_t ws_size,
                              hipStream_t stream) {
  using namespace hgrn;
  (void)in_sizes; (void)n_in; (void)out_size; (void)ws_size;

  const float* x  = (const float*)d_in[0];
  const float* Wp[4] = { (const float*)d_in[1], (const float*)d_in[2],
                         (const float*)d_in[3], (const float*)d_in[4] };
  const float* gw = (const float*)d_in[5];
  float* out = (float*)d_out;

  char* ws = (char*)d_ws;
  float*  wsum   = (float*)(ws + OFF_WSUM);
  float*  wpart  = (float*)(ws + OFF_WPART);
  float*  xscale = (float*)(ws + OFF_XSCL);
  float*  oscale = (float*)(ws + OFF_OSCL);
  int8_t* wq     = (int8_t*)(ws + OFF_WQ);
  int8_t* xq     = (int8_t*)(ws + OFF_XQ);
  int8_t* oq     = (int8_t*)(ws + OFF_OQ);
  float*  iraw   = (float*)(ws + OFF_IRAW);
  float*  fraw   = (float*)(ws + OFF_FRAW);
  float*  graw   = (float*)(ws + OFF_GRAW);
  float*  csP    = (float*)(ws + OFF_CSP);
  float*  csH    = (float*)(ws + OFF_CSH);
  float*  carry  = (float*)(ws + OFF_CARRY);

  for (int w = 0; w < 4; ++w)
    wabs_partial_kernel<<<256, 256, 0, stream>>>(Wp[w], wpart + (size_t)w * 256);
  wsum_final_kernel<<<4, 256, 0, stream>>>(wpart, wsum);
  for (int w = 0; w < 4; ++w)
    wquant_kernel<<<2048, 256, 0, stream>>>(Wp[w], wsum + w, wq + (size_t)w * DD);

  act_quant_kernel<<<M, 256, 0, stream>>>(x, xq, xscale);

  dim3 gg(Nd / 128, M / 128);
  gemm_iu8_kernel<<<gg, 256, 0, stream>>>(xq, wq + 0 * DD, xscale, wsum + 0, iraw);
  gemm_iu8_kernel<<<gg, 256, 0, stream>>>(xq, wq + 1 * DD, xscale, wsum + 1, fraw);
  gemm_iu8_kernel<<<gg, 256, 0, stream>>>(xq, wq + 2 * DD, xscale, wsum + 2, graw);

  swiglu_kernel<<<8192, 256, 0, stream>>>(iraw, fraw);
  scan_pass1_kernel<<<(CH * BD) / 256, 256, 0, stream>>>(iraw, fraw, csP, csH);
  scan_pass2_kernel<<<BD / 256, 256, 0, stream>>>(csP, csH, carry);
  scan_pass3_kernel<<<8192, 256, 0, stream>>>(iraw, fraw, carry);

  gate_quant_kernel<<<M, 256, 0, stream>>>(graw, iraw, gw, oq, oscale);

  gemm_iu8_kernel<<<gg, 256, 0, stream>>>(oq, wq + 3 * DD, oscale, wsum + 3, out);
}
